// CapAttention_40183714021653
// MI455X (gfx1250) — compile-verified, matrix operand order checked by reading, not run on hardware
//
#include <hip/hip_runtime.h>
#include <hip/hip_bf16.h>
#include <cstdint>
#include <cstddef>

// ---------------- problem constants ----------------
#define B_   32
#define S_   512
#define D_   640
#define H_   10
#define DK_  64
#define N_   (B_ * S_)          // 16384 tokens

typedef __attribute__((ext_vector_type(16))) __bf16 v16bf;
typedef __attribute__((ext_vector_type(8)))  float  v8f;

__device__ __forceinline__ unsigned short f32_to_bf16_bits(float f) {
  union { float f; unsigned u; } v; v.f = f;
  unsigned u = v.u;
  return (unsigned short)((u + 0x7FFFu + ((u >> 16) & 1u)) >> 16);
}

__device__ __forceinline__ v8f wmma_bf16(v16bf a, v16bf b, v8f c) {
  // (neg_a, A, neg_b, B, c_mod, C, reuse_a, reuse_b)
  return __builtin_amdgcn_wmma_f32_16x16x32_bf16(false, a, false, b, (short)0, c,
                                                 false, false);
}

// ---------------- elementwise casts ----------------
__global__ __launch_bounds__(256) void cap_cast_f32_bf16(const float* __restrict__ src,
                                                         unsigned short* __restrict__ dst,
                                                         int n) {
  int i = blockIdx.x * 256 + threadIdx.x;
  if (i < n) dst[i] = f32_to_bf16_bits(src[i]);
}

// route_weights [1,H,DK,D] f32 -> rwT [H,D,DK] bf16 (B-columns contiguous)
__global__ __launch_bounds__(256) void cap_cast_rw_T(const float* __restrict__ rw,
                                                     unsigned short* __restrict__ rwT) {
  int idx = blockIdx.x * 256 + threadIdx.x;            // over H*DK*D
  if (idx >= H_ * DK_ * D_) return;
  int o = idx % D_;
  int i = (idx / D_) % DK_;
  int r = idx / (D_ * DK_);
  rwT[((size_t)r * D_ + o) * DK_ + i] = f32_to_bf16_bits(rw[idx]);
}

// ---------------- projection GEMM: out = X @ W^T + b ----------------
// X: [N,640] bf16, W: [640,640] bf16 (row o == B-column o), bias f32.
// vtrans==0 -> out [B,H,S,DK] bf16 ; vtrans==1 -> out [B,H,DK,S] bf16 (V^T)
__global__ __launch_bounds__(256) void cap_proj_kernel(const __bf16* __restrict__ X,
                                                       const __bf16* __restrict__ W,
                                                       const float* __restrict__ bias,
                                                       unsigned short* __restrict__ out,
                                                       int vtrans) {
  const int NCT = D_ / 16;                 // 40 column tiles
  int wid  = threadIdx.x >> 5;
  int lane = threadIdx.x & 31;
  int tile = blockIdx.x * 8 + wid;         // (N/16)*40 tiles total
  int rowt = tile / NCT;
  int colt = tile % NCT;
  int m  = lane & 15;
  int kh = lane >> 4;                      // which 16-wide K sub-strip

  const __bf16* arow = X + (size_t)(rowt * 16 + m) * D_ + kh * 16;
  const __bf16* brow = W + (size_t)(colt * 16 + m) * D_ + kh * 16;

  v8f acc = {};
#pragma unroll
  for (int kc = 0; kc < D_ / 32; ++kc) {   // 20 K-chunks of 32
    v16bf a = *(const v16bf*)(arow + kc * 32);
    v16bf b = *(const v16bf*)(brow + kc * 32);
    acc = wmma_bf16(a, b, acc);
  }

  int ncol = colt * 16 + m;                // output column o
  float bv = bias[ncol];
  int h  = ncol >> 6;
  int dk = ncol & 63;
#pragma unroll
  for (int j = 0; j < 8; ++j) {
    int tok = rowt * 16 + j + (kh << 3);   // lanes >=16 hold M = j+8
    int bb = tok >> 9;
    int ss = tok & 511;
    unsigned short val = f32_to_bf16_bits(acc[j] + bv);
    size_t addr;
    if (!vtrans) addr = (((size_t)(bb * H_ + h) * S_) + ss) * DK_ + dk;
    else         addr = (((size_t)(bb * H_ + h) * DK_) + dk) * S_ + ss;
    out[addr] = val;
  }
}

// ---------------- attention: scores, softmax, P@V ----------------
// block = 128 threads = 4 waves, each wave owns a 16-row q tile.
// dynamic LDS: float sc[4][16][512] (128 KB) + __bf16 pb[4][16][512] (64 KB)
__global__ __launch_bounds__(128) void cap_attn_kernel(const __bf16* __restrict__ Q,
                                                       const __bf16* __restrict__ K,
                                                       const __bf16* __restrict__ VT,
                                                       const float* __restrict__ adj,
                                                       const int* __restrict__ mask,
                                                       const int* __restrict__ layer_p,
                                                       unsigned short* __restrict__ Xout) {
  extern __shared__ char smem_attn[];
  int wid  = threadIdx.x >> 5;
  int lane = threadIdx.x & 31;
  int bid  = blockIdx.x;                   // B*H*8 blocks
  int qb   = bid & 7;
  int bh   = bid >> 3;
  int h    = bh % H_;
  int b    = bh / H_;
  int q0   = qb * 64 + wid * 16;

  float*  sc = (float*)smem_attn + (size_t)wid * 16 * S_;               // f32 scores
  __bf16* pb = (__bf16*)(smem_attn + 4 * 16 * S_ * sizeof(float)) +
               (size_t)wid * 16 * S_;                                   // bf16 probs
  int m  = lane & 15;
  int kh = lane >> 4;

  // A fragments for this wave's 16 q rows (K = 64 -> two 32-chunks)
  const __bf16* qbase = Q + ((size_t)(b * H_ + h) * S_ + q0 + m) * DK_ + kh * 16;
  v16bf a0 = *(const v16bf*)(qbase);
  v16bf a1 = *(const v16bf*)(qbase + 32);

  const __bf16* kmat = K + (size_t)(b * H_ + h) * S_ * DK_;
  for (int kt = 0; kt < S_ / 16; ++kt) {   // 32 key tiles
    const __bf16* kb = kmat + (size_t)(kt * 16 + m) * DK_ + kh * 16;
    v16bf b0 = *(const v16bf*)(kb);
    v16bf b1 = *(const v16bf*)(kb + 32);
    v8f acc = {};
    acc = wmma_bf16(a0, b0, acc);
    acc = wmma_bf16(a1, b1, acc);
    int n = kt * 16 + m;
#pragma unroll
    for (int j = 0; j < 8; ++j)
      sc[(size_t)(j + (kh << 3)) * S_ + n] = acc[j];
  }

  // ---- softmax over 512 keys; lane handles (row = lane&15, half = lane>>4) ----
  int row  = lane & 15;
  int half = lane >> 4;
  float inv_layer = 1.0f / (float)(layer_p[0] + 1);
  const float* adjrow  = adj  + ((size_t)b * S_ + q0 + row) * S_ + half * 256;
  const int*   maskrow = mask + ((size_t)b * S_ + q0 + row) * S_ + half * 256;
  float*    srow = sc + (size_t)row * S_ + half * 256;
  unsigned* prow_pack = (unsigned*)(pb + (size_t)row * S_ + half * 256);

  float mx = -3.0e38f;
  for (int j = 0; j < 256; ++j) {
    float v = srow[j] * 0.125f + adjrow[j] * inv_layer;  // 1/sqrt(64)
    if (maskrow[j] == 0) v = -1.0e9f;
    srow[j] = v;
    mx = fmaxf(mx, v);
  }
  mx = fmaxf(mx, __shfl_xor(mx, 16, 32));
  float sum = 0.0f;
  for (int j = 0; j < 256; ++j) {
    float e = __expf(srow[j] - mx);
    srow[j] = e;
    sum += e;
  }
  sum += __shfl_xor(sum, 16, 32);
  float inv = 1.0f / sum;
  // normalized probabilities packed to bf16 (two per ds_store_b32)
  for (int j = 0; j < 256; j += 2) {
    unsigned lo = f32_to_bf16_bits(srow[j] * inv);
    unsigned hi = f32_to_bf16_bits(srow[j + 1] * inv);
    prow_pack[j >> 1] = lo | (hi << 16);
  }

  // ---- out[16x64] = P[16x512] @ V[512x64] ----
  v8f oacc[4] = {v8f{}, v8f{}, v8f{}, v8f{}};
  const __bf16* vmat = VT + (size_t)(b * H_ + h) * DK_ * S_;
  for (int c = 0; c < S_ / 32; ++c) {      // 16 K-chunks
    v16bf a = *(const v16bf*)(pb + (size_t)(lane & 15) * S_ + c * 32 + kh * 16);
#pragma unroll
    for (int nt = 0; nt < 4; ++nt) {
      v16bf bf = *(const v16bf*)(vmat + (size_t)(nt * 16 + (lane & 15)) * S_ +
                                 c * 32 + kh * 16);
      oacc[nt] = wmma_bf16(a, bf, oacc[nt]);
    }
  }

  // store token-major x: [N, H, DK] bf16
#pragma unroll
  for (int nt = 0; nt < 4; ++nt) {
    int dk = nt * 16 + (lane & 15);
#pragma unroll
    for (int j = 0; j < 8; ++j) {
      int tok = b * S_ + q0 + j + (kh << 3);
      Xout[((size_t)tok * H_ + h) * DK_ + dk] = f32_to_bf16_bits(oacc[nt][j]);
    }
  }
}

// ---------------- fused priors GEMM + capsule routing ----------------
// block = 256 threads (8 waves) handles 16 tokens.
// dynamic LDS: priors bf16 [10][16][640] (204800B) + obuf f32 [16][640] (40960B)
//            + logits [160] + red [160] + snb [16]  => 247104 B
__global__ __launch_bounds__(256) void cap_route_kernel(const __bf16* __restrict__ X,
                                                        const __bf16* __restrict__ RWT,
                                                        float* __restrict__ out) {
  extern __shared__ char smem_route[];
  __bf16* priors = (__bf16*)smem_route;                         // [10][16][640]
  float*  obuf   = (float*)(smem_route + 204800);               // [16][640]
  float*  logits = (float*)(smem_route + 204800 + 40960);       // [16][10]
  float*  red    = logits + 160;                                // [16][10]
  float*  snb    = red + 160;                                   // [16]

  int wid  = threadIdx.x >> 5;
  int lane = threadIdx.x & 31;
  int tok0 = blockIdx.x * 16;
  int m  = lane & 15;
  int kh = lane >> 4;

  // priors[r] = x[:,r,:] (16x64) @ rw[r] (64x640), bf16 into LDS
  for (int r = 0; r < H_; ++r) {
    const __bf16* abase = X + ((size_t)(tok0 + m) * H_ + r) * DK_ + kh * 16;
    v16bf a0 = *(const v16bf*)(abase);
    v16bf a1 = *(const v16bf*)(abase + 32);
    for (int nt = wid; nt < D_ / 16; nt += 8) {    // 40 tiles / 8 waves
      const __bf16* bbase = RWT + ((size_t)r * D_ + nt * 16 + m) * DK_ + kh * 16;
      v16bf b0 = *(const v16bf*)(bbase);
      v16bf b1 = *(const v16bf*)(bbase + 32);
      v8f acc = {};
      acc = wmma_bf16(a0, b0, acc);
      acc = wmma_bf16(a1, b1, acc);
      int o = nt * 16 + m;
#pragma unroll
      for (int j = 0; j < 8; ++j) {
        int t = j + (kh << 3);
        priors[((size_t)r * 16 + t) * D_ + o] = (__bf16)acc[j];
      }
    }
  }
  if (threadIdx.x < 160) logits[threadIdx.x] = 0.0f;
  __syncthreads();

  int tg = threadIdx.x >> 4;   // token 0..15
  int tl = threadIdx.x & 15;   // worker within token group

  for (int it = 0; it < 3; ++it) {
    // softmax over route nodes (each of 16 threads computes the same; cheap)
    float probs[H_];
    float mx = -3.0e38f;
    for (int r = 0; r < H_; ++r) mx = fmaxf(mx, logits[tg * H_ + r]);
    float s = 0.0f;
    for (int r = 0; r < H_; ++r) { probs[r] = __expf(logits[tg * H_ + r] - mx); s += probs[r]; }
    float pinv = 1.0f / s;
    for (int r = 0; r < H_; ++r) probs[r] *= pinv;

    if (tl == 0) snb[tg] = 0.0f;
    if (tl < H_) red[tg * H_ + tl] = 0.0f;
    __syncthreads();

    // out = sum_r probs_r * priors_r ; accumulate squared norm
    float psn = 0.0f;
    for (int o = tl; o < D_; o += 16) {
      float acc = 0.0f;
#pragma unroll
      for (int r = 0; r < H_; ++r)
        acc += probs[r] * (float)priors[((size_t)r * 16 + tg) * D_ + o];
      obuf[tg * D_ + o] = acc;
      psn += acc * acc;
    }
    atomicAdd(&snb[tg], psn);
    __syncthreads();

    float sn = snb[tg];
    float scale = (sn / (1.0f + sn)) * __frsqrt_rn(sn + 1e-9f);  // squash

    if (it < 2) {
      float pd[H_];
#pragma unroll
      for (int r = 0; r < H_; ++r) pd[r] = 0.0f;
      for (int o = tl; o < D_; o += 16) {
        float ov = obuf[tg * D_ + o] * scale;
#pragma unroll
        for (int r = 0; r < H_; ++r)
          pd[r] += (float)priors[((size_t)r * 16 + tg) * D_ + o] * ov;
      }
#pragma unroll
      for (int r = 0; r < H_; ++r) atomicAdd(&red[tg * H_ + r], pd[r]);
      __syncthreads();
      if (tl == 0)
        for (int r = 0; r < H_; ++r) logits[tg * H_ + r] += red[tg * H_ + r];
      __syncthreads();
    } else {
      for (int o = tl; o < D_; o += 16)
        out[(size_t)(tok0 + tg) * D_ + o] = obuf[tg * D_ + o] * scale;
    }
  }
}

// ---------------- host launcher ----------------
extern "C" void kernel_launch(void* const* d_in, const int* in_sizes, int n_in,
                              void* d_out, int out_size, void* d_ws, size_t ws_size,
                              hipStream_t stream) {
  const float* query = (const float*)d_in[0];
  const float* key   = (const float*)d_in[1];
  const float* value = (const float*)d_in[2];
  const int*   mask  = (const int*)d_in[3];
  const float* adj   = (const float*)d_in[4];
  const int*   layer = (const int*)d_in[5];
  const float* Wq    = (const float*)d_in[6];
  const float* bq    = (const float*)d_in[7];
  const float* Wk    = (const float*)d_in[8];
  const float* bk    = (const float*)d_in[9];
  const float* Wv    = (const float*)d_in[10];
  const float* bv    = (const float*)d_in[11];
  const float* rw    = (const float*)d_in[12];

  char* ws = (char*)d_ws;
  size_t off = 0;
  auto carve = [&](size_t bytes) {
    void* p = ws + off;
    off += (bytes + 255) & ~(size_t)255;
    return p;
  };
  const size_t XB = (size_t)N_ * D_ * 2;      // 20,971,520 B
  const size_t WB = (size_t)D_ * D_ * 2;      //    819,200 B
  unsigned short* xq_b  = (unsigned short*)carve(XB);
  unsigned short* xk_b  = (unsigned short*)carve(XB);
  unsigned short* xv_b  = (unsigned short*)carve(XB);
  unsigned short* wq_b  = (unsigned short*)carve(WB);
  unsigned short* wk_b  = (unsigned short*)carve(WB);
  unsigned short* wv_b  = (unsigned short*)carve(WB);
  unsigned short* rwT_b = (unsigned short*)carve((size_t)H_ * D_ * DK_ * 2);
  unsigned short* Qh    = (unsigned short*)carve(XB);   // [B,H,S,DK]
  unsigned short* Kh    = (unsigned short*)carve(XB);   // [B,H,S,DK]
  unsigned short* VTh   = (unsigned short*)carve(XB);   // [B,H,DK,S]
  unsigned short* Xtok  = (unsigned short*)carve(XB);   // [N,H,DK]

  // 1) casts
  const int nX = N_ * D_;
  const int nW = D_ * D_;
  cap_cast_f32_bf16<<<(nX + 255) / 256, 256, 0, stream>>>(query, xq_b, nX);
  cap_cast_f32_bf16<<<(nX + 255) / 256, 256, 0, stream>>>(key,   xk_b, nX);
  cap_cast_f32_bf16<<<(nX + 255) / 256, 256, 0, stream>>>(value, xv_b, nX);
  cap_cast_f32_bf16<<<(nW + 255) / 256, 256, 0, stream>>>(Wq, wq_b, nW);
  cap_cast_f32_bf16<<<(nW + 255) / 256, 256, 0, stream>>>(Wk, wk_b, nW);
  cap_cast_f32_bf16<<<(nW + 255) / 256, 256, 0, stream>>>(Wv, wv_b, nW);
  cap_cast_rw_T<<<(H_ * DK_ * D_ + 255) / 256, 256, 0, stream>>>(rw, rwT_b);

  // 2) projections: (N/16)*(D/16) = 40960 tiles, 8 tiles/block
  const int projBlocks = (N_ / 16) * (D_ / 16) / 8;    // 5120
  cap_proj_kernel<<<projBlocks, 256, 0, stream>>>((const __bf16*)xq_b, (const __bf16*)wq_b, bq, Qh, 0);
  cap_proj_kernel<<<projBlocks, 256, 0, stream>>>((const __bf16*)xk_b, (const __bf16*)wk_b, bk, Kh, 0);
  cap_proj_kernel<<<projBlocks, 256, 0, stream>>>((const __bf16*)xv_b, (const __bf16*)wv_b, bv, VTh, 1);

  // 3) attention: B*H*(S/64) blocks of 4 waves, 192 KB dynamic LDS
  const int attnBlocks = B_ * H_ * (S_ / 64);          // 2560
  const size_t attnLds = (size_t)4 * 16 * S_ * sizeof(float)     // f32 scores
                       + (size_t)4 * 16 * S_ * sizeof(short);    // bf16 probs
  cap_attn_kernel<<<attnBlocks, 128, attnLds, stream>>>(
      (const __bf16*)Qh, (const __bf16*)Kh, (const __bf16*)VTh,
      adj, mask, layer, Xtok);

  // 4) priors + routing: N/16 blocks, 247104 B dynamic LDS
  const int routeBlocks = N_ / 16;                     // 1024
  const size_t routeLds = 204800 + 40960 + 160 * 4 + 160 * 4 + 16 * 4;
  cap_route_kernel<<<routeBlocks, 256, routeLds, stream>>>(
      (const __bf16*)Xtok, (const __bf16*)rwT_b, (float*)d_out);
}
